// PointNetSetAbstractionMsg_24111946399910
// MI455X (gfx1250) — matrix operand order repaired and checked by reading, not compiled, not run on hardware
//
#include <hip/hip_runtime.h>
#include <hip/hip_bf16.h>

#define B_ 16
#define N_ 2048
#define D_ 64
#define S_ 512
#define CIN_ (D_ + 3)
#define CPAD0_ 96            // first-layer input channels padded 67 -> 96
#define TW 2                 // 16-wide col tiles per wave

typedef __attribute__((ext_vector_type(16))) __bf16 v16bf;
typedef __attribute__((ext_vector_type(8)))  __bf16 v8bf;
typedef __attribute__((ext_vector_type(8)))  float  v8f;

// ---------------------------------------------------------------------------
// 1) Farthest point sampling: one block per batch, 512 sequential argmax steps.
// ---------------------------------------------------------------------------
__global__ void fps_kernel(const float* __restrict__ xyz, float* __restrict__ out)
{
    const int b = blockIdx.x;
    const int t = threadIdx.x;
    const float* xb = xyz + (size_t)b * 3 * N_;

    float px[8], py[8], pz[8], dist[8];
#pragma unroll
    for (int i = 0; i < 8; ++i) {
        int n = t + i * 256;
        px[i] = xb[n];
        py[i] = xb[N_ + n];
        pz[i] = xb[2 * N_ + n];
        dist[i] = 1e10f;
    }

    __shared__ float s_val[256];
    __shared__ int   s_idx[256];
    __shared__ float s_c[3];

    int far = 0;
    for (int it = 0; it < S_; ++it) {
        if (t == 0) {
            float cx = xb[far], cy = xb[N_ + far], cz = xb[2 * N_ + far];
            s_c[0] = cx; s_c[1] = cy; s_c[2] = cz;
            out[(size_t)b * 3 * S_ + it]          = cx;
            out[(size_t)b * 3 * S_ + S_ + it]     = cy;
            out[(size_t)b * 3 * S_ + 2 * S_ + it] = cz;
        }
        __syncthreads();
        float cx = s_c[0], cy = s_c[1], cz = s_c[2];
        float best = -1.f; int bi = 0;
#pragma unroll
        for (int i = 0; i < 8; ++i) {
            float dx = px[i] - cx, dy = py[i] - cy, dz = pz[i] - cz;
            float d = dx * dx + dy * dy + dz * dz;
            float nd = fminf(dist[i], d);
            dist[i] = nd;
            if (nd > best) { best = nd; bi = t + i * 256; }
        }
        s_val[t] = best; s_idx[t] = bi;
        __syncthreads();
        for (int off = 128; off > 0; off >>= 1) {
            if (t < off) {
                if (s_val[t + off] > s_val[t]) {
                    s_val[t] = s_val[t + off];
                    s_idx[t] = s_idx[t + off];
                }
            }
            __syncthreads();
        }
        far = s_idx[0];
    }
}

// ---------------------------------------------------------------------------
// 2) Ball query: one wave32 per query, ballot + prefix-popcount append.
// ---------------------------------------------------------------------------
__global__ void ball_query_kernel(const float* __restrict__ xyz,
                                  const float* __restrict__ newxyz, // (B,3,S) in d_out
                                  int* __restrict__ idx, float r2, int K)
{
    const int wave = threadIdx.x >> 5;
    const int lane = threadIdx.x & 31;
    const int q = blockIdx.x * (blockDim.x >> 5) + wave;   // q in [0, B*S)
    const int b = q / S_;
    const int s = q - b * S_;

    const float cx = newxyz[(size_t)b * 3 * S_ + s];
    const float cy = newxyz[(size_t)b * 3 * S_ + S_ + s];
    const float cz = newxyz[(size_t)b * 3 * S_ + 2 * S_ + s];
    const float* xb = xyz + (size_t)b * 3 * N_;
    int* outq = idx + (size_t)q * K;

    int count = 0;
    int first = 0;
    for (int base = 0; base < N_ && count < K; base += 32) {
        int n = base + lane;
        float dx = xb[n] - cx, dy = xb[N_ + n] - cy, dz = xb[2 * N_ + n] - cz;
        bool pred = (dx * dx + dy * dy + dz * dz <= r2);
        unsigned m = (unsigned)__ballot(pred);
        if (count == 0 && m) first = base + __ffs(m) - 1;
        int pos = count + __popc(m & ((1u << lane) - 1u));
        if (pred && pos < K) outq[pos] = n;
        count += __popc(m);
    }
    if (count < K) {
        for (int j = count + lane; j < K; j += 32) outq[j] = first;
    }
}

// ---------------------------------------------------------------------------
// 3) Gather + concat -> bf16 X[col][Cpad], col = (b*S+s)*K + k.
// ---------------------------------------------------------------------------
__global__ void gather_kernel(const float* __restrict__ xyz,
                              const float* __restrict__ points,
                              const float* __restrict__ newxyz,
                              const int* __restrict__ idx,
                              __bf16* __restrict__ X, int K, long long cols, int Cpad)
{
    long long tid = (long long)blockIdx.x * blockDim.x + threadIdx.x;
    if (tid >= (long long)cols * Cpad) return;
    int c = (int)(tid % Cpad);
    long long col = tid / Cpad;
    int q = (int)(col / K);           // b*S + s
    int b = q / S_;
    int s = q - b * S_;
    float v = 0.f;
    if (c < D_) {
        int id = idx[col];
        v = points[((size_t)b * D_ + c) * N_ + id];
    } else if (c < CIN_) {
        int id = idx[col];
        int cc = c - D_;
        v = xyz[((size_t)b * 3 + cc) * N_ + id]
          - newxyz[((size_t)b * 3 + cc) * S_ + s];
    }
    X[(size_t)col * Cpad + c] = (__bf16)v;
}

// ---------------------------------------------------------------------------
// fp32 [OC][Csrc] weights -> zero-padded bf16 [OC][Cdst].
// ---------------------------------------------------------------------------
__global__ void convert_w_kernel(const float* __restrict__ Wf, __bf16* __restrict__ Wb,
                                 int Csrc, int Cdst, int OC)
{
    int t = blockIdx.x * blockDim.x + threadIdx.x;
    if (t >= OC * Cdst) return;
    int o = t / Cdst, c = t - o * Cdst;
    Wb[t] = (c < Csrc) ? (__bf16)Wf[o * Csrc + c] : (__bf16)0.f;
}

__global__ void zero_stats_kernel(float* a, float* b)
{
    int t = threadIdx.x;
    if (t < 128) { a[t] = 0.f; b[t] = 0.f; }
}

// ---------------------------------------------------------------------------
// 4) bf16 WMMA GEMM, full-OC per wave (ROWT = OC/16 row tiles).
//    Per K-step: load TW B-fragments once, reuse across all ROWT WMMAs.
//    AFFINE: apply previous layer's BN scale/shift + ReLU to the B fragment
//    on load (VALU co-executes with TRANS-class WMMA); X streamed once.
//    Stats (sum/sumsq of raw pre-BN output) accumulated in the epilogue.
// ---------------------------------------------------------------------------
template<int ROWT, bool AFFINE>
__global__ void gemm_wmma_kernel(const __bf16* __restrict__ Wb,   // [OC][C]
                                 const __bf16* __restrict__ X,    // [cols][C]
                                 __bf16* __restrict__ Y,          // [cols][OC]
                                 const float* __restrict__ pscl,  // prev-layer scale
                                 const float* __restrict__ pshf,  // prev-layer shift
                                 float* __restrict__ sum, float* __restrict__ sumsq,
                                 int C, long long cols)
{
    const int OC = ROWT * 16;
    const int wave = threadIdx.x >> 5;
    const int lane = threadIdx.x & 31;
    const int mn = lane & 15;                // M for A, N (col) for B/C/D
    const int hi = lane >> 4;                // half-wave select
    const long long colBase = ((long long)blockIdx.x * 8 + wave) * (16 * TW);

    // A: lane needs W[r*16+mn][k + hi*16 .. +15]
    const __bf16* wrow = Wb + (size_t)mn * C + hi * 16;
    // B: lane needs X[col][k + hi*16 .. +15] (one contiguous 32B chunk)
    const __bf16* xcol[TW];
#pragma unroll
    for (int t = 0; t < TW; ++t)
        xcol[t] = X + (size_t)(colBase + t * 16 + mn) * C + hi * 16;

    v8f zero = {};
    v8f acc[ROWT][TW];
#pragma unroll
    for (int r = 0; r < ROWT; ++r)
#pragma unroll
        for (int t = 0; t < TW; ++t) acc[r][t] = zero;

    for (int k = 0; k < C; k += 32) {
        v16bf bfrag[TW];
#pragma unroll
        for (int t = 0; t < TW; ++t)
            bfrag[t] = *(const v16bf*)(xcol[t] + k);
        if (AFFINE) {
            // channels of bfrag element i: k + hi*16 + i (uniform per half-wave)
            const float* scp = pscl + k + hi * 16;
            const float* shp = pshf + k + hi * 16;
#pragma unroll
            for (int t = 0; t < TW; ++t)
#pragma unroll
                for (int i = 0; i < 16; ++i) {
                    float v = (float)bfrag[t][i] * scp[i] + shp[i];
                    bfrag[t][i] = (__bf16)fmaxf(v, 0.f);
                }
        }
#pragma unroll
        for (int r = 0; r < ROWT; ++r) {
            const __bf16* wr = wrow + (size_t)r * 16 * C + k;
            v8bf a0 = *(const v8bf*)(wr);
            v8bf a1 = *(const v8bf*)(wr + 8);
            v16bf a = __builtin_shufflevector(a0, a1,
                        0, 1, 2, 3, 4, 5, 6, 7, 8, 9, 10, 11, 12, 13, 14, 15);
#pragma unroll
            for (int t = 0; t < TW; ++t)
                acc[r][t] = __builtin_amdgcn_wmma_f32_16x16x32_bf16(
                                false, a, false, bfrag[t], (short)0, acc[r][t],
                                false, false);
        }
    }

    // Epilogue. C/D layout: VGPR j holds row r*16 + hi*8 + j, col colBase+t*16+mn.
#pragma unroll
    for (int r = 0; r < ROWT; ++r) {
        float s1[8], s2[8];
#pragma unroll
        for (int j = 0; j < 8; ++j) { s1[j] = 0.f; s2[j] = 0.f; }
#pragma unroll
        for (int t = 0; t < TW; ++t) {
            v8bf yo;
#pragma unroll
            for (int j = 0; j < 8; ++j) {
                float v = acc[r][t][j];
                yo[j] = (__bf16)v;
                s1[j] += v;
                s2[j] += v * v;
            }
            *(v8bf*)(Y + (size_t)(colBase + t * 16 + mn) * OC + r * 16 + hi * 8) = yo;
        }
#pragma unroll
        for (int j = 0; j < 8; ++j) {
            float a = s1[j], b = s2[j];
#pragma unroll
            for (int off = 1; off < 16; off <<= 1) {    // stays inside 16-lane half
                a += __shfl_xor(a, off, 32);
                b += __shfl_xor(b, off, 32);
            }
            if (mn == 0) {
                int rr = r * 16 + hi * 8 + j;
                atomicAdd(&sum[rr], a);
                atomicAdd(&sumsq[rr], b);
            }
        }
    }
}

// ---------------------------------------------------------------------------
// 5) BN finalize; fused BN+ReLU+max-over-K final epilogue.
// ---------------------------------------------------------------------------
__global__ void bn_finalize_kernel(const float* __restrict__ sum,
                                   const float* __restrict__ sumsq,
                                   const float* __restrict__ g,
                                   const float* __restrict__ beta,
                                   float* __restrict__ scl, float* __restrict__ shf,
                                   int OC, float invcnt)
{
    int o = threadIdx.x;
    if (o >= OC) return;
    float m  = sum[o] * invcnt;
    float v  = sumsq[o] * invcnt - m * m;
    float sc = g[o] * rsqrtf(v + 1e-5f);
    scl[o] = sc;
    shf[o] = beta[o] - m * sc;
}

// Fused: out[b, chOff+o, s] = max_k relu(Y[(b*S+s)*K+k][o] * scl + shf)
__global__ void maxk_kernel(const __bf16* __restrict__ Y,
                            const float* __restrict__ scl,
                            const float* __restrict__ shf,
                            float* __restrict__ out,
                            int OC, int K, int chOff)
{
    int tid = blockIdx.x * blockDim.x + threadIdx.x;
    int groups = OC >> 3;
    if (tid >= B_ * S_ * groups) return;
    int g = tid % groups;
    int o0 = g * 8;
    int rest = tid / groups;
    int s = rest % S_;
    int b = rest / S_;

    float sc[8], sh[8], m[8];
#pragma unroll
    for (int i = 0; i < 8; ++i) { sc[i] = scl[o0 + i]; sh[i] = shf[o0 + i]; m[i] = 0.f; }

    const __bf16* y = Y + (size_t)(b * S_ + s) * K * OC + o0;
    for (int k = 0; k < K; ++k) {
        v8bf yv = *(const v8bf*)(y + (size_t)k * OC);
#pragma unroll
        for (int i = 0; i < 8; ++i) {
            float v = fmaxf((float)yv[i] * sc[i] + sh[i], 0.f);
            m[i] = fmaxf(m[i], v);
        }
    }
    size_t ob = (size_t)B_ * 3 * S_ + ((size_t)b * 320 + chOff + o0) * S_ + s;
#pragma unroll
    for (int i = 0; i < 8; ++i) out[ob + (size_t)i * S_] = m[i];
}

// ---------------------------------------------------------------------------
// Host orchestration.
// ---------------------------------------------------------------------------
extern "C" void kernel_launch(void* const* d_in, const int* in_sizes, int n_in,
                              void* d_out, int out_size, void* d_ws, size_t ws_size,
                              hipStream_t stream)
{
    const float* xyz    = (const float*)d_in[0];
    const float* points = (const float*)d_in[1];
    float* out = (float*)d_out;

    const int   mlps[3][3] = {{32, 32, 64}, {64, 64, 128}, {64, 96, 128}};
    const int   nsample[3] = {16, 32, 128};
    const float radius[3]  = {0.1f, 0.2f, 0.4f};
    const int   choff[3]   = {0, 64, 192};

    // workspace carve-up
    char* w = (char*)d_ws;
    int* idx = (int*)w;                 w += (size_t)B_ * S_ * 128 * sizeof(int);
    float* sum   = (float*)w;           w += 128 * sizeof(float);
    float* sumsq = (float*)w;           w += 128 * sizeof(float);
    float* scl   = (float*)w;           w += 128 * sizeof(float);
    float* shf   = (float*)w;           w += 128 * sizeof(float);
    __bf16* Wb   = (__bf16*)w;          w += 16384 * sizeof(__bf16);
    size_t off = (size_t)(w - (char*)d_ws);
    off = (off + 255) & ~(size_t)255;
    const size_t bufElems = (size_t)B_ * S_ * 128 * 128;   // 1M cols x 128 ch
    __bf16* bufA = (__bf16*)((char*)d_ws + off);
    __bf16* bufB = bufA + bufElems;

    // 1) FPS -> new_xyz into d_out
    fps_kernel<<<B_, 256, 0, stream>>>(xyz, out);

    for (int sc = 0; sc < 3; ++sc) {
        const int K = nsample[sc];
        const long long cols = (long long)B_ * S_ * K;

        // 2) ball query
        ball_query_kernel<<<(B_ * S_) / 8, 256, 0, stream>>>(
            xyz, out, idx, radius[sc] * radius[sc], K);

        // 3) gather + concat -> bufA [cols][CPAD0_]
        long long gtot = cols * CPAD0_;
        gather_kernel<<<(unsigned)((gtot + 255) / 256), 256, 0, stream>>>(
            xyz, points, out, idx, bufA, K, cols, CPAD0_);

        int Csrc = CIN_;     // logical input channels of layer 0
        int Cpad = CPAD0_;   // padded (storage) channels
        __bf16* X = bufA;
        __bf16* Y = bufB;
        for (int j = 0; j < 3; ++j) {
            const int OC = mlps[sc][j];
            const int base = 2 + (sc * 3 + j) * 4;
            const float* Wf   = (const float*)d_in[base + 0];
            const float* g    = (const float*)d_in[base + 2];
            const float* beta = (const float*)d_in[base + 3];

            convert_w_kernel<<<(OC * Cpad + 255) / 256, 256, 0, stream>>>(
                Wf, Wb, Csrc, Cpad, OC);
            zero_stats_kernel<<<1, 128, 0, stream>>>(sum, sumsq);

            // 8 waves x TW x 16 = 256 cols per block; cols % 256 == 0 always
            dim3 gg((unsigned)(cols / (16 * TW * 8)), 1);
            const int rt = OC / 16;
            if (j == 0) {   // raw gathered input, no affine
                switch (rt) {
                case 2:  gemm_wmma_kernel<2, false><<<gg, 256, 0, stream>>>(Wb, X, Y, scl, shf, sum, sumsq, Cpad, cols); break;
                default: gemm_wmma_kernel<4, false><<<gg, 256, 0, stream>>>(Wb, X, Y, scl, shf, sum, sumsq, Cpad, cols); break;
                }
            } else {        // apply previous layer's BN+ReLU on load
                switch (rt) {
                case 2:  gemm_wmma_kernel<2, true><<<gg, 256, 0, stream>>>(Wb, X, Y, scl, shf, sum, sumsq, Cpad, cols); break;
                case 4:  gemm_wmma_kernel<4, true><<<gg, 256, 0, stream>>>(Wb, X, Y, scl, shf, sum, sumsq, Cpad, cols); break;
                case 6:  gemm_wmma_kernel<6, true><<<gg, 256, 0, stream>>>(Wb, X, Y, scl, shf, sum, sumsq, Cpad, cols); break;
                default: gemm_wmma_kernel<8, true><<<gg, 256, 0, stream>>>(Wb, X, Y, scl, shf, sum, sumsq, Cpad, cols); break;
                }
            }

            // finalize AFTER the GEMM that consumed the previous scl/shf
            bn_finalize_kernel<<<1, 128, 0, stream>>>(sum, sumsq, g, beta, scl, shf, OC,
                                                      1.0f / (float)cols);

            __bf16* t = X; X = Y; Y = t;   // ping-pong
            Csrc = OC;
            Cpad = OC;                     // subsequent layers: already % 32 == 0
        }

        // 4) fused BN+ReLU+max over K (final raw GEMM output is in X after 3 swaps)
        const int OCl = mlps[sc][2];
        int mtot = B_ * S_ * (OCl / 8);
        maxk_kernel<<<(mtot + 255) / 256, 256, 0, stream>>>(
            X, scl, shf, out, OCl, K, choff[sc]);
    }
}
